// VectorQuantizer_59734405153101
// MI455X (gfx1250) — matrix operand order repaired and checked by reading, not compiled, not run on hardware
//
#include <hip/hip_runtime.h>

typedef __attribute__((ext_vector_type(16))) _Float16 v16h;
typedef __attribute__((ext_vector_type(8)))  float    v8f;

// ---------------------------------------------------------------------------
// Prep 1: bias[k] = -0.5 * ||W_k||^2   (f32, K entries)
// ---------------------------------------------------------------------------
__global__ void vq_prep_bias(const float* __restrict__ W,
                             float* __restrict__ bias, int K) {
  int k = blockIdx.x * blockDim.x + threadIdx.x;
  if (k < K) {
    const float* wr = W + (size_t)k * 64;
    float s = 0.f;
#pragma unroll
    for (int d = 0; d < 64; ++d) s += wr[d] * wr[d];
    bias[k] = -0.5f * s;
  }
}

// ---------------------------------------------------------------------------
// Prep 2: repack W (K x 64 f32) into f16 WMMA B-matrix lane layout.
// Contiguous k order k = 16h + j, matching A (16-bit B, 32x16): tile t (16 codes), K-chunk c (d base c*32):
//   lane L: column n = L%16 -> code k = t*16+n
//           K base = (L<16 ? 0 : 16); half j (0..15) -> d = c*32 + Kbase + j
// Stored contiguously per lane: Bbuf[(((t*2)+c)*32 + L)*16 + j]
// ---------------------------------------------------------------------------
__global__ void vq_prep_b(const float* __restrict__ W,
                          _Float16* __restrict__ Bbuf, int K) {
  int idx = blockIdx.x * blockDim.x + threadIdx.x;
  int total = K * 64;
  if (idx >= total) return;
  int j = idx & 15;
  int L = (idx >> 4) & 31;
  int c = (idx >> 9) & 1;
  int t = idx >> 10;
  int n = L & 15;
  int kb = (L < 16) ? 0 : 16;
  int k = t * 16 + n;
  int d = c * 32 + kb + j;
  Bbuf[idx] = (_Float16)W[(size_t)k * 64 + d];
}

// ---------------------------------------------------------------------------
// Main kernel.
//  * 256 threads = 8 wave32; block covers 128 z-rows; each wave a 16-row tile.
//  * B streamed through LDS in double-buffered 4-tile (8 KB) stages, shared by
//    all 8 waves; per stage each thread prefetches 32 B of stage s+1 into
//    registers BEFORE computing and commits to LDS AFTER computing.
//  * WMMA accumulators seed from inline-0 SRC2; bias (-0.5||w||^2) is added at
//    compare time so the bias load is off the WMMA critical path.
// ---------------------------------------------------------------------------
__global__ __launch_bounds__(256) void vq_main(
    const float* __restrict__ z, const float* __restrict__ W,
    const _Float16* __restrict__ Bbuf, const float* __restrict__ bias,
    float* __restrict__ out, int nRows, int K) {
  __shared__ __align__(32) float    zl[128 * 68];       // 34,816 B
  __shared__ __align__(32) _Float16 Bst[2][4 * 1024];   // 16,384 B (2 x 4 tiles)
  __shared__ __align__(32) float    biasl[1024];        //  4,096 B
  const int tid = threadIdx.x;
  const int R0 = blockIdx.x * 128;

  // ---- stage z tile (coalesced float4 reads) ----
  for (int i = tid; i < 128 * 16; i += 256) {
    int row = i >> 4, c4 = i & 15;
    int gr = R0 + row;
    if (gr >= nRows) gr = nRows - 1;
    const float4 v = ((const float4*)(z + (size_t)gr * 64))[c4];
    float* dst = &zl[row * 68 + c4 * 4];
    dst[0] = v.x; dst[1] = v.y; dst[2] = v.z; dst[3] = v.w;
  }
  // ---- stage bias (once) ----
  for (int i = tid; i < K && i < 1024; i += 256) biasl[i] = bias[i];
  // ---- stage B buffer 0 (tiles 0..3) ----
  *(v16h*)(&Bst[0][tid * 16]) = *(const v16h*)(Bbuf + (size_t)tid * 16);
  __syncthreads();

  const int wave = tid >> 5;
  const int lane = tid & 31;
  const int lm   = lane & 15;
  const int cb   = (lane < 16) ? 0 : 16;  // A-matrix K sub-base (contiguous k = 16h + s)
  const float* zrow = &zl[(wave * 16 + lm) * 68];

  // ---- build A (16x64 f16) once: two 16x32 chunks in contiguous k order (k = 16h + s) ----
  v16h a0, a1;
#pragma unroll
  for (int j = 0; j < 8; ++j) {
    a0[j]     = (_Float16)zrow[cb + j];            // K = cb + j
    a0[8 + j] = (_Float16)zrow[cb + 8 + j];        // K = cb + 8 + j
    a1[j]     = (_Float16)zrow[32 + cb + j];
    a1[8 + j] = (_Float16)zrow[40 + cb + j];
  }

  float best[8];
  int   bidx[8];
#pragma unroll
  for (int r = 0; r < 8; ++r) { best[r] = -3.0e38f; bidx[r] = 0; }

  const int tiles   = K >> 4;        // 64 code tiles
  const int nStages = tiles >> 2;    // 16 stages of 4 tiles

  for (int s = 0; s < nStages; ++s) {
    const int cur = s & 1, nxt = cur ^ 1;

    // -- issue global prefetch of stage s+1 into registers (uniform clamp) --
    const int sn = (s + 1 < nStages) ? (s + 1) : 0;
    v16h pre = *(const v16h*)(Bbuf + (size_t)sn * 4096 + (size_t)tid * 16);

    // -- compute 4 tiles from current LDS stage --
#pragma unroll
    for (int u = 0; u < 4; ++u) {
      const int t = s * 4 + u;
      const _Float16* bp = &Bst[cur][u * 1024 + lane * 16];
      v16h b0 = *(const v16h*)bp;            // K-chunk 0 (d 0..31)
      v16h b1 = *(const v16h*)(bp + 512);    // K-chunk 1 (d 32..63)
      float bb = biasl[t * 16 + lm];         // only needed at compare time

      v8f c = {};  // inline-0 SRC2: WMMA issues as soon as B arrives
      c = __builtin_amdgcn_wmma_f32_16x16x32_f16(false, a0, false, b0,
                                                 (short)0, c, false, false);
      c = __builtin_amdgcn_wmma_f32_16x16x32_f16(false, a1, false, b1,
                                                 (short)0, c, false, false);

      const int kn = t * 16 + lm;  // code index held by this lane's column
#pragma unroll
      for (int r = 0; r < 8; ++r) {
        float v = c[r] + bb;       // score = z.w - 0.5||w||^2
        if (v > best[r]) { best[r] = v; bidx[r] = kn; }  // lowest k wins ties
      }
    }

    // -- commit prefetched stage to the other LDS buffer, then sync --
    *(v16h*)(&Bst[nxt][tid * 16]) = pre;
    __syncthreads();
  }

  // ---- argmax across the 16 lanes of each half (rows m<8 in lanes 0-15,
  //      rows m>=8 in lanes 16-31), first-index tie-break ----
#pragma unroll
  for (int mask = 1; mask <= 8; mask <<= 1) {
#pragma unroll
    for (int r = 0; r < 8; ++r) {
      float ov = __shfl_xor(best[r], mask, 32);
      int   oi = __shfl_xor(bidx[r], mask, 32);
      if (ov > best[r] || (ov == best[r] && oi < bidx[r])) {
        best[r] = ov; bidx[r] = oi;
      }
    }
  }

  // ---- emit 2*W[k] - z for the wave's 16 rows (coalesced float2 stores) ----
#pragma unroll
  for (int m = 0; m < 16; ++m) {
    const int src = (m < 8) ? 0 : 16;
    const int r = m & 7;
    int k = __shfl(bidx[r], src, 32);
    int gr = R0 + wave * 16 + m;
    if (gr < nRows) {
      float2 wv = *(const float2*)(W + (size_t)k * 64 + lane * 2);
      const float* zr = &zl[(wave * 16 + m) * 68 + lane * 2];
      float2 o;
      o.x = 2.f * wv.x - zr[0];
      o.y = 2.f * wv.y - zr[1];
      *(float2*)(out + (size_t)gr * 64 + lane * 2) = o;
    }
  }
}

// ---------------------------------------------------------------------------
extern "C" void kernel_launch(void* const* d_in, const int* in_sizes, int n_in,
                              void* d_out, int out_size, void* d_ws, size_t ws_size,
                              hipStream_t stream) {
  const float* z = (const float*)d_in[0];
  const float* W = (const float*)d_in[1];
  const int nZ = in_sizes[0];        // 256*1024*64 = 16,777,216
  const int KD = in_sizes[1];        // 1024*64
  const int K = KD / 64;             // 1024 codes
  const int nRows = nZ / 64;         // 262,144 rows

  float*    bias = (float*)d_ws;                         // K floats (4 KB)
  _Float16* Bbuf = (_Float16*)((char*)d_ws + 4096);      // K*64 halves (128 KB)

  vq_prep_bias<<<(K + 255) / 256, 256, 0, stream>>>(W, bias, K);
  vq_prep_b<<<(K * 64 + 255) / 256, 256, 0, stream>>>(W, Bbuf, K);

  const int blocks = (nRows + 127) / 128;
  vq_main<<<blocks, 256, 0, stream>>>(z, W, Bbuf, bias, (float*)d_out,
                                      nRows, K);
}
